// GNN_627065225439
// MI455X (gfx1250) — matrix operand order repaired and checked by reading, not compiled, not run on hardware
//
#include <hip/hip_runtime.h>
#include <hip/hip_bf16.h>

typedef __attribute__((ext_vector_type(16))) _Float16 v16h;
typedef __attribute__((ext_vector_type(8)))  float    v8f;

#define DD   128   // feature dim
#define DH   256   // hidden dim

// Wave-private LDS staging only needs compiler-level ordering: the LDS pipe is
// in-order per wave, so cross-lane data written by earlier ds ops is visible to
// later ds ops of the same wave without a workgroup barrier.
#if defined(__has_builtin)
#if __has_builtin(__builtin_amdgcn_wave_barrier)
#define WAVE_FENCE() __builtin_amdgcn_wave_barrier()
#else
#define WAVE_FENCE() __syncthreads()
#endif
#else
#define WAVE_FENCE() __syncthreads()
#endif

// ---------------------------------------------------------------------------
// One-time: convert W1/W2 (f32 row-major [layer][k][n]) into f16 B-fragment
// layout for v_wmma_f32_16x16x32_f16.
// B-frag (K=32 x N=16): lane l holds column n = ntile*16 + (l&15);
// half h (0..15) holds K = chunk*32 + (l>>4)*8 + h (h<8), +16+(h-8) (h>=8).
// Storage: frag(ntile,chunk) = 32 lanes x 16 halves = 1KB contiguous.
// ---------------------------------------------------------------------------
__global__ void swizzle_weights(const float* __restrict__ W1,
                                const float* __restrict__ W2,
                                _Float16* __restrict__ w1f,
                                _Float16* __restrict__ w2f, int L)
{
    int gid = blockIdx.x * blockDim.x + threadIdx.x;
    if (gid >= L * 128 * 32) return;          // L * (64+64 frags) * 32 lanes
    int lane  = gid & 31;
    int frag  = (gid >> 5) & 127;             // 0..127
    int layer = (gid >> 5) >> 7;              // 0..L-1
    int lg = lane >> 4, nl = lane & 15;

    const float* W; _Float16* out; int ntile, chunk, ncols;
    if (frag < 64) {                          // W1: 16 ntiles x 4 chunks
        ntile = frag >> 2; chunk = frag & 3;
        W = W1 + (size_t)layer * DD * DH; ncols = DH;
        out = w1f + (((size_t)layer * 64 + frag) * 32 + lane) * 16;
    } else {                                  // W2: 8 ntiles x 8 chunks
        int f = frag - 64;
        ntile = f >> 3; chunk = f & 7;
        W = W2 + (size_t)layer * DH * DD; ncols = DD;
        out = w2f + (((size_t)layer * 64 + f) * 32 + lane) * 16;
    }
    int n  = ntile * 16 + nl;
    int kb = chunk * 32 + lg * 8;
#pragma unroll
    for (int h = 0; h < 16; ++h)
        out[h] = (_Float16)W[(size_t)(kb + h + (h & 8)) * ncols + n];
}

// ---------------------------------------------------------------------------
// z = (1 + eps[layer]) * h      (float4 vectorized)
// ---------------------------------------------------------------------------
__global__ void scale_h(const float* __restrict__ h, float* __restrict__ z,
                        const float* __restrict__ eps, int layer, int total4)
{
    int i = blockIdx.x * blockDim.x + threadIdx.x;
    if (i >= total4) return;
    float e = 1.0f + eps[layer];
    float4 v = ((const float4*)h)[i];
    v.x *= e; v.y *= e; v.z *= e; v.w *= e;
    ((float4*)z)[i] = v;
}

// ---------------------------------------------------------------------------
// Edge scatter-add: one wave per edge, 32 lanes x float4 = 128 dims.
// ---------------------------------------------------------------------------
__global__ void scatter_edges(const float* __restrict__ h,
                              const int* __restrict__ src,
                              const int* __restrict__ dst,
                              float* __restrict__ z, int E)
{
    int wv   = (int)((blockIdx.x * (unsigned)blockDim.x + threadIdx.x) >> 5);
    int lane = threadIdx.x & 31;
    if (wv >= E) return;
    int s = src[wv], d = dst[wv];
    float4 v = *(const float4*)(h + (size_t)s * DD + lane * 4);
    float* zp = z + (size_t)d * DD + lane * 4;
    atomicAdd(zp + 0, v.x); atomicAdd(zp + 1, v.y);
    atomicAdd(zp + 2, v.z); atomicAdd(zp + 3, v.w);
}

// ---------------------------------------------------------------------------
// Fused GIN MLP: hout = [relu]( relu(z @ W1 + b1) @ W2 + b2 )
// One wave computes a 16-row tile; 8 waves (256 thr) per block = 128 rows.
// A-frag (16-bit A 16x32): lane l row m=l&15; halves 0..7 = K kb..kb+7,
// halves 8..15 = K kb+16..kb+23, kb = 32c + (l>>4)*8.
// ---------------------------------------------------------------------------
__global__ __launch_bounds__(256) void gin_mlp(
    const float* __restrict__ z,
    const _Float16* __restrict__ w1f, const _Float16* __restrict__ w2f,
    const float* __restrict__ b1, const float* __restrict__ b2,
    float* __restrict__ hout, int layer, int relu_out, int Nnodes)
{
    __shared__ float stg[8][16 * 16];
    const int wave = threadIdx.x >> 5;
    const int lane = threadIdx.x & 31;
    const int lg   = lane >> 4;
    const int nl   = lane & 15;
    const long tile = (long)blockIdx.x * 8 + wave;
    const long row0 = tile * 16;
    if (row0 >= Nnodes) return;               // wave-uniform exit

    // ---- build A-fragments of z (16 x 128 f32 -> 4 x v16h) ----
    v16h a[4];
    {
        const float* zr = z + (row0 + nl) * DD;
#pragma unroll
        for (int c = 0; c < 4; ++c) {
            int kb = c * 32 + lg * 8;
            float4 p0 = *(const float4*)(zr + kb);
            float4 p1 = *(const float4*)(zr + kb + 4);
            float4 p2 = *(const float4*)(zr + kb + 16);
            float4 p3 = *(const float4*)(zr + kb + 20);
            a[c][0]  = (_Float16)p0.x; a[c][1]  = (_Float16)p0.y;
            a[c][2]  = (_Float16)p0.z; a[c][3]  = (_Float16)p0.w;
            a[c][4]  = (_Float16)p1.x; a[c][5]  = (_Float16)p1.y;
            a[c][6]  = (_Float16)p1.z; a[c][7]  = (_Float16)p1.w;
            a[c][8]  = (_Float16)p2.x; a[c][9]  = (_Float16)p2.y;
            a[c][10] = (_Float16)p2.z; a[c][11] = (_Float16)p2.w;
            a[c][12] = (_Float16)p3.x; a[c][13] = (_Float16)p3.y;
            a[c][14] = (_Float16)p3.z; a[c][15] = (_Float16)p3.w;
        }
    }

    // ---- GEMM1: hidden(16x256) = relu(z @ W1 + b1), kept as A-frags ----
    const float* b1l = b1 + (size_t)layer * DH;
    v16h hfrag[8];
#pragma unroll
    for (int t = 0; t < 16; ++t) {
        v8f acc = {};
        const _Float16* bp = w1f + (((size_t)layer * 64 + t * 4) * 32 + lane) * 16;
#pragma unroll
        for (int c = 0; c < 4; ++c) {
            v16h b = *(const v16h*)(bp + (size_t)c * 32 * 16);
            acc = __builtin_amdgcn_wmma_f32_16x16x32_f16(
                false, a[c], false, b, (short)0, acc, false, false);
        }
        float bias = b1l[t * 16 + nl];
#pragma unroll
        for (int v = 0; v < 8; ++v) {
            float x = acc[v] + bias;                       // C elem (m=v+8*lg, n=nl)
            stg[wave][(v + 8 * lg) * 16 + nl] = x > 0.f ? x : 0.f;
        }
        WAVE_FENCE();   // order ds stores before ds loads (in-order LDS per wave)
        // read back as A-frag halves: lane -> row m=nl, cols lg*8+0..7
        const int c2 = t >> 1;
        const int hb = (t & 1) ? 8 : 0;
        const float* sp = &stg[wave][nl * 16 + lg * 8];
#pragma unroll
        for (int i = 0; i < 8; ++i)
            hfrag[c2][hb + i] = (_Float16)sp[i];
        WAVE_FENCE();   // keep next iteration's stores after these loads
    }

    // ---- GEMM2: out(16x128) = hidden @ W2 + b2 (+optional relu) ----
    const float* b2l = b2 + (size_t)layer * DD;
    float* hr = hout + row0 * DD;
#pragma unroll
    for (int o = 0; o < 8; ++o) {
        v8f acc = {};
        const _Float16* bp = w2f + (((size_t)layer * 64 + o * 8) * 32 + lane) * 16;
#pragma unroll
        for (int c = 0; c < 8; ++c) {
            v16h b = *(const v16h*)(bp + (size_t)c * 32 * 16);
            acc = __builtin_amdgcn_wmma_f32_16x16x32_f16(
                false, hfrag[c], false, b, (short)0, acc, false, false);
        }
        float bias = b2l[o * 16 + nl];
#pragma unroll
        for (int v = 0; v < 8; ++v) {
            float x = acc[v] + bias;
            if (relu_out) x = x > 0.f ? x : 0.f;
            hr[(size_t)(v + 8 * lg) * DD + o * 16 + nl] = x;
        }
    }
}

// ---------------------------------------------------------------------------
// Pooling helpers
// ---------------------------------------------------------------------------
__global__ void compute_offsets(const int* __restrict__ nsg,
                                int* __restrict__ offs, int G)
{
    if (blockIdx.x == 0 && threadIdx.x == 0) {
        int acc = 0;
        for (int g = 0; g < G; ++g) { offs[g] = acc; acc += nsg[g]; }
    }
}

__global__ void zero_f32(float* __restrict__ p, int n)
{
    int i = blockIdx.x * blockDim.x + threadIdx.x;
    if (i < n) p[i] = 0.0f;
}

__global__ void pool_sum(const float* __restrict__ h,
                         const int* __restrict__ batch,
                         const int* __restrict__ sgb,
                         const int* __restrict__ offs,
                         float* __restrict__ out, float* __restrict__ cnt, int N)
{
    int wv   = (int)((blockIdx.x * (unsigned)blockDim.x + threadIdx.x) >> 5);
    int lane = threadIdx.x & 31;
    if (wv >= N) return;
    int sg = offs[batch[wv]] + sgb[wv];
    float4 v = *(const float4*)(h + (size_t)wv * DD + lane * 4);
    float* op = out + (size_t)sg * DD + lane * 4;
    atomicAdd(op + 0, v.x); atomicAdd(op + 1, v.y);
    atomicAdd(op + 2, v.z); atomicAdd(op + 3, v.w);
    if (lane == 0) atomicAdd(cnt + sg, 1.0f);
}

__global__ void pool_div(float* __restrict__ out, const float* __restrict__ cnt,
                         int total)
{
    int i = blockIdx.x * blockDim.x + threadIdx.x;
    if (i >= total) return;
    float c = cnt[i >> 7];
    out[i] /= (c > 1.0f ? c : 1.0f);
}

// ---------------------------------------------------------------------------
extern "C" void kernel_launch(void* const* d_in, const int* in_sizes, int n_in,
                              void* d_out, int out_size, void* d_ws, size_t ws_size,
                              hipStream_t stream)
{
    const float* x    = (const float*)d_in[0];
    const int*   ei   = (const int*)  d_in[1];
    const int*   batch= (const int*)  d_in[2];
    const int*   sgb  = (const int*)  d_in[3];
    const int*   nsg  = (const int*)  d_in[4];
    const float* W1   = (const float*)d_in[5];
    const float* b1   = (const float*)d_in[6];
    const float* W2   = (const float*)d_in[7];
    const float* b2   = (const float*)d_in[8];
    const float* eps  = (const float*)d_in[9];

    const int N = in_sizes[0] / DD;
    const int E = in_sizes[1] / 2;
    const int G = in_sizes[4];
    const int L = in_sizes[9];
    float* out = (float*)d_out;

    char* ws = (char*)d_ws;
    size_t off = 0;
    float*    buf0 = (float*)(ws + off);    off += (size_t)N * DD * 4;
    float*    buf1 = (float*)(ws + off);    off += (size_t)N * DD * 4;
    _Float16* w1f  = (_Float16*)(ws + off); off += (size_t)L * 64 * 32 * 16 * 2;
    _Float16* w2f  = (_Float16*)(ws + off); off += (size_t)L * 64 * 32 * 16 * 2;
    int*      offs = (int*)(ws + off);      off += 256;
    float*    cnt  = (float*)(ws + off);    off += 8192;
    (void)ws_size; (void)n_in;

    // one-time prep
    {
        int total = L * 128 * 32;
        swizzle_weights<<<(total + 255) / 256, 256, 0, stream>>>(W1, W2, w1f, w2f, L);
    }
    compute_offsets<<<1, 32, 0, stream>>>(nsg, offs, G);

    const int* src = ei;
    const int* dst = ei + E;
    for (int l = 0; l < L; ++l) {
        const float* hin = (l == 0) ? x : buf0;
        int total4 = N * (DD / 4);
        scale_h<<<(total4 + 255) / 256, 256, 0, stream>>>(hin, buf1, eps, l, total4);
        scatter_edges<<<(E + 7) / 8, 256, 0, stream>>>(hin, src, dst, buf1, E);
        int tiles = (N + 15) / 16;
        gin_mlp<<<(tiles + 7) / 8, 256, 0, stream>>>(buf1, w1f, w2f, b1, b2,
                                                     buf0, l, (l < L - 1) ? 1 : 0, N);
    }

    int nsgT = out_size / DD;
    zero_f32<<<(out_size + 255) / 256, 256, 0, stream>>>(out, out_size);
    zero_f32<<<(nsgT + 255) / 256, 256, 0, stream>>>(cnt, nsgT);
    pool_sum<<<(N + 7) / 8, 256, 0, stream>>>(buf0, batch, sgb, offs, out, cnt, N);
    pool_div<<<(out_size + 255) / 256, 256, 0, stream>>>(out, cnt, out_size);
}
